// VAEEncoder_66760971649072
// MI455X (gfx1250) — matrix operand order repaired and checked by reading, not compiled, not run on hardware
//
#include <hip/hip_runtime.h>
#include <hip/hip_bf16.h>

typedef __attribute__((ext_vector_type(16))) _Float16 v16h;
typedef __attribute__((ext_vector_type(8)))  float    v8f;
typedef __attribute__((ext_vector_type(4)))  unsigned int u32x4;
typedef __attribute__((ext_vector_type(4)))  int i32x4;
typedef __attribute__((ext_vector_type(8)))  int i32x8;

#define N_NODES  131072
#define N_EDGES  524288
#define N_TYPES  4
#define DIMD     128
#define DIMM     256
#define N_GRAPH  4096
#define DIM_GH   512
#define DIM_L    128
#define N_PASSES 4

// ---------------------------------------------------------------------------
// Generic f32 -> f16 conversion
// ---------------------------------------------------------------------------
__global__ void f32_to_f16_kernel(const float* __restrict__ s,
                                  _Float16* __restrict__ d, int n) {
  int i = blockIdx.x * blockDim.x + threadIdx.x;
  if (i < n) d[i] = (_Float16)s[i];
}

// ---------------------------------------------------------------------------
// h = emb[node_types]  (f32 + f16 copies)
// ---------------------------------------------------------------------------
__global__ void embed_kernel(const int* __restrict__ types,
                             const float* __restrict__ emb,
                             float* __restrict__ h, _Float16* __restrict__ h16) {
  int i = blockIdx.x * blockDim.x + threadIdx.x;      // over N*D, exact multiple
  int n = i >> 7, d = i & 127;
  float v = emb[(size_t)types[n] * DIMD + d];
  h[i] = v;
  h16[i] = (_Float16)v;
}

// ---------------------------------------------------------------------------
// TDM tile load: 64 rows x 16 dwords (64B) tile, row stride Kd dwords,
// into LDS with 1-dword padding after every 16 dwords (=> 17-dword LDS rows).
// D# per CDNA5 ISA §8.3-8.5: group0 {count=1, lds_addr, global_addr, type=2},
// group1 {data_size=4B, pad_enable, pad_interval=3(16dw), pad_amount=0(1dw),
//         tensor_dim0=Kd, tensor_dim1=nrows, tile_dim0=16, tile_dim1=64,
//         tensor_dim0_stride=Kd}.  Groups 2/3 zero (2-D tensor).
// ---------------------------------------------------------------------------
__device__ __forceinline__ void tdm_load_tile(unsigned lds_addr,
                                              const _Float16* gsrc,
                                              int Kd, int nrows)
{
  unsigned long long ga = (unsigned long long)(size_t)gsrc;
  u32x4 g0;
  g0[0] = 1u;                                            // count=1
  g0[1] = lds_addr;                                      // LDS byte address
  g0[2] = (unsigned)(ga & 0xFFFFFFFFu);                  // global_addr[31:0]
  g0[3] = (unsigned)((ga >> 32) & 0x1FFFFFFu)            // global_addr[56:32]
        | (2u << 30);                                    // type=2 ("image")

  unsigned long long q0 = (2ull << 16)                   // data_size: 4B
                        | (1ull << 20)                   // pad_enable
                        | (3ull << 22)                   // pad_interval: 16 dwords
                        /* pad_amount=0 -> 1 dword */
                        | ((unsigned long long)((unsigned)Kd & 0xFFFFu) << 48);
  unsigned long long q1 = ((unsigned long long)(((unsigned)Kd >> 16) & 0xFFFFu))
                        | ((unsigned long long)(unsigned)nrows << 16)  // tensor_dim1
                        | (16ull << 48);                 // tile_dim0 = 16 dwords
  unsigned long long q2 = 64ull                          // tile_dim1 = 64 rows
                        | ((unsigned long long)(unsigned)Kd << 32); // dim0_stride
  unsigned long long q3 = 0ull;                          // dim1_stride unused

  i32x8 g1;
  g1[0] = (int)(unsigned)q0; g1[1] = (int)(q0 >> 32);
  g1[2] = (int)(unsigned)q1; g1[3] = (int)(q1 >> 32);
  g1[4] = (int)(unsigned)q2; g1[5] = (int)(q2 >> 32);
  g1[6] = (int)(unsigned)q3; g1[7] = (int)(q3 >> 32);

  i32x4 z4 = {0, 0, 0, 0};
#if __clang_major__ >= 23
  i32x8 z8 = {0, 0, 0, 0, 0, 0, 0, 0};
  __builtin_amdgcn_tensor_load_to_lds(g0, g1, z4, z4, z8, 0);
#else
  __builtin_amdgcn_tensor_load_to_lds(g0, g1, z4, z4, 0);
#endif
}

// ---------------------------------------------------------------------------
// WMMA GEMM:  C(rows x ncols) = A(rows x K) * B(ncols x K)^T + bias
//   A, B f16 row-major; C f32. Block = 128 threads = 4 waves.
//   Block tile 64x64; wave w owns rows [16w,16w+16) x 64 cols (4 N-tiles).
//   A/B tiles staged to LDS by the Tensor Data Mover (wave 0 issues).
//   mode 0: C[r*ncols+c] = acc + bias[c]
//   mode 1: atomicAdd(C[seg[r]*ncols+c], gate[r]*(acc+bias[c]))   (readout)
// ---------------------------------------------------------------------------
__global__ void __launch_bounds__(128)
gemm_wmma_kernel(const _Float16* __restrict__ A, const _Float16* __restrict__ B,
                 const float* __restrict__ bias, float* __restrict__ C,
                 int K, int ncols, int mode,
                 const int* __restrict__ seg, const float* __restrict__ gate)
{
  __shared__ unsigned int As_u[64 * 17];   // 64 rows x 16 dwords + 1 pad dword
  __shared__ unsigned int Bs_u[64 * 17];
  const _Float16* Ash = (const _Float16*)As_u;   // element (r,k) at [r*34 + k]
  const _Float16* Bsh = (const _Float16*)Bs_u;

  const int rowBase = blockIdx.y * 64;
  const int colBase = blockIdx.x * 64;
  const int nrowsA  = (int)gridDim.y * 64;
  const int tid  = threadIdx.x;
  const int w    = tid >> 5;        // wave id 0..3 (wave32)
  const int lane = tid & 31;
  const int hi16 = lane >> 4;       // lane half
  const int l16  = lane & 15;

  const unsigned ldsA = (unsigned)(size_t)&As_u[0];
  const unsigned ldsB = (unsigned)(size_t)&Bs_u[0];

  v8f acc[4] = {};
  const int Kd = K >> 1;            // row length in dwords

  for (int k0 = 0; k0 < K; k0 += 32) {
    if (tid < 32) {                 // wave 0 drives the TDM (EXEC ignored by TDM)
      tdm_load_tile(ldsA, A + (size_t)rowBase * K + k0, Kd, nrowsA);
      tdm_load_tile(ldsB, B + (size_t)colBase * K + k0, Kd, ncols);
      __builtin_amdgcn_s_wait_tensorcnt(0);
    }
    __syncthreads();

    // A fragment (16x32): lanes 0-15 -> M=l16, K = {2j | j<4} U {16+2j}; lanes 16-31: K+8
    v16h af;
#pragma unroll
    for (int j = 0; j < 8; ++j) {
      int kb = ((j < 4) ? (2 * j) : (16 + 2 * (j - 4))) + 8 * hi16;
      const _Float16* p = &Ash[(16 * w + l16) * 34 + kb];
      af[2 * j]     = p[0];
      af[2 * j + 1] = p[1];
    }
#pragma unroll
    for (int nt = 0; nt < 4; ++nt) {
      // B fragment (32x16): same K order as A: lanes 0-15 K = {2j | j<4} U {16+2j}; lanes 16-31: K+8
      v16h bf;
#pragma unroll
      for (int j = 0; j < 8; ++j) {
        const _Float16* p = &Bsh[(16 * nt + l16) * 34 + ((j < 4) ? (2 * j) : (16 + 2 * (j - 4))) + 8 * hi16];
        bf[2 * j]     = p[0];
        bf[2 * j + 1] = p[1];
      }
      acc[nt] = __builtin_amdgcn_wmma_f32_16x16x32_f16(
          /*neg_a=*/false, af, /*neg_b=*/false, bf,
          /*c_mod=*/(short)0, acc[nt], /*reuse_a=*/false, /*reuse_b=*/false);
    }
    __syncthreads();
  }

  // Epilogue. C/D layout: VGPR i -> M = i + 8*laneHalf, N = l16.
#pragma unroll
  for (int nt = 0; nt < 4; ++nt) {
    int c = colBase + 16 * nt + l16;
    float bv = bias ? bias[c] : 0.f;
#pragma unroll
    for (int i = 0; i < 8; ++i) {
      int r = rowBase + 16 * w + 8 * hi16 + i;
      float v = acc[nt][i] + bv;
      if (mode == 0) {
        C[(size_t)r * ncols + c] = v;
      } else {
        atomicAdd(&C[(size_t)seg[r] * ncols + c], gate[r] * v);
      }
    }
  }
}

// ---------------------------------------------------------------------------
// agg[dst] += msgs[src]  for edges of type t (4 floats per thread)
// ---------------------------------------------------------------------------
__global__ void scatter_edges_kernel(const float* __restrict__ msgs,
                                     const int* __restrict__ esrc,
                                     const int* __restrict__ edst,
                                     const int* __restrict__ etype,
                                     int t, float* __restrict__ agg)
{
  int tid = blockIdx.x * blockDim.x + threadIdx.x;  // E * 64 threads
  int e = tid >> 6;
  int c = (tid & 63) << 2;
  if (etype[e] != t) return;
  int s = esrc[e], d0 = edst[e];
  const float4 mv = *(const float4*)&msgs[(size_t)s * DIMM + c];
  float* ap = &agg[(size_t)d0 * DIMM + c];
  atomicAdd(ap + 0, mv.x);
  atomicAdd(ap + 1, mv.y);
  atomicAdd(ap + 2, mv.z);
  atomicAdd(ap + 3, mv.w);
}

// ---------------------------------------------------------------------------
// m16 = f16(relu(agg))
// ---------------------------------------------------------------------------
__global__ void relu_to_f16_kernel(const float* __restrict__ a,
                                   _Float16* __restrict__ o, int n) {
  int i = blockIdx.x * blockDim.x + threadIdx.x;
  if (i < n) {
    float v = a[i];
    o[i] = (_Float16)(v > 0.f ? v : 0.f);
  }
}

// ---------------------------------------------------------------------------
// GRU update (elementwise):  h = (1-z)*n + z*h
// ---------------------------------------------------------------------------
__global__ void gru_kernel(const float* __restrict__ gi,
                           const float* __restrict__ gh,
                           float* __restrict__ h, _Float16* __restrict__ h16) {
  int i = blockIdx.x * blockDim.x + threadIdx.x;   // N*D
  int n = i >> 7, d = i & 127;
  size_t b = (size_t)n * 384;
  float ir = gi[b + d], iz = gi[b + 128 + d], in = gi[b + 256 + d];
  float hr = gh[b + d], hz = gh[b + 128 + d], hn = gh[b + 256 + d];
  float r  = 1.f / (1.f + __expf(-(ir + hr)));
  float z  = 1.f / (1.f + __expf(-(iz + hz)));
  float nn = tanhf(in + r * hn);
  float hv = (1.f - z) * nn + z * h[i];
  h[i]   = hv;
  h16[i] = (_Float16)hv;
}

// ---------------------------------------------------------------------------
// gate[n] = sigmoid(h[n].W_gate + b);  seg[n] = searchsorted(ptr,n,'right')-1
// ---------------------------------------------------------------------------
__global__ void gate_seg_kernel(const float* __restrict__ h,
                                const float* __restrict__ Wg,
                                const float* __restrict__ bg,
                                const int* __restrict__ ptr,
                                float* __restrict__ gate, int* __restrict__ seg) {
  int n = blockIdx.x * blockDim.x + threadIdx.x;
  if (n >= N_NODES) return;
  float s = bg[0];
  const float* hp = &h[(size_t)n * DIMD];
#pragma unroll 8
  for (int d = 0; d < DIMD; ++d) s += hp[d] * Wg[d];
  gate[n] = 1.f / (1.f + __expf(-s));
  // largest l in [0, G] with ptr[l] <= n
  int l = 0, r = N_GRAPH;
  while (l < r) {
    int m = (l + r + 1) >> 1;
    if (ptr[m] <= n) l = m; else r = m - 1;
  }
  seg[n] = l;
}

// ---------------------------------------------------------------------------
extern "C" void kernel_launch(void* const* d_in, const int* in_sizes, int n_in,
                              void* d_out, int out_size, void* d_ws, size_t ws_size,
                              hipStream_t stream)
{
  const int*   node_types = (const int*)d_in[0];
  const int*   edge_src   = (const int*)d_in[1];
  const int*   edge_dst   = (const int*)d_in[2];
  const int*   edge_type  = (const int*)d_in[3];
  const int*   ptr        = (const int*)d_in[4];
  const float* emb    = (const float*)d_in[5];
  const float* W_msg  = (const float*)d_in[6];
  const float* b_msg  = (const float*)d_in[7];
  const float* W_ih   = (const float*)d_in[8];
  const float* W_hh   = (const float*)d_in[9];
  const float* b_ih   = (const float*)d_in[10];
  const float* b_hh   = (const float*)d_in[11];
  const float* W_gate = (const float*)d_in[12];
  const float* b_gate = (const float*)d_in[13];
  const float* W_n2g  = (const float*)d_in[14];
  const float* b_n2g  = (const float*)d_in[15];
  const float* W_loc  = (const float*)d_in[16];
  const float* b_loc  = (const float*)d_in[17];
  const float* W_lv   = (const float*)d_in[18];
  const float* b_lv   = (const float*)d_in[19];
  float* out = (float*)d_out;
  (void)in_sizes; (void)n_in; (void)out_size; (void)ws_size;

  char* ws = (char*)d_ws;
  size_t off = 0;
  auto alloc = [&](size_t bytes) -> char* {
    char* p = ws + off;
    off += (bytes + 255) & ~(size_t)255;
    return p;
  };
  float*    h      = (float*)   alloc((size_t)N_NODES * DIMD * 4);
  _Float16* h16    = (_Float16*)alloc((size_t)N_NODES * DIMD * 2);
  float*    msgs   = (float*)   alloc((size_t)N_NODES * DIMM * 4);   // one type at a time
  float*    agg    = (float*)   alloc((size_t)N_NODES * DIMM * 4);
  _Float16* m16    = (_Float16*)alloc((size_t)N_NODES * DIMM * 2);
  float*    gi     = (float*)   alloc((size_t)N_NODES * 384 * 4);
  float*    gh     = (float*)   alloc((size_t)N_NODES * 384 * 4);
  float*    gate   = (float*)   alloc((size_t)N_NODES * 4);
  int*      seg    = (int*)     alloc((size_t)N_NODES * 4);
  float*    hgraph = (float*)   alloc((size_t)N_GRAPH * DIM_GH * 4);
  _Float16* hg16   = (_Float16*)alloc((size_t)N_GRAPH * DIM_GH * 2);
  _Float16* Wmsg16 = (_Float16*)alloc((size_t)N_TYPES * DIMM * DIMD * 2);
  _Float16* Wih16  = (_Float16*)alloc((size_t)384 * DIMM * 2);
  _Float16* Whh16  = (_Float16*)alloc((size_t)384 * DIMD * 2);
  _Float16* Wn2g16 = (_Float16*)alloc((size_t)DIM_GH * DIMD * 2);
  _Float16* Wloc16 = (_Float16*)alloc((size_t)DIM_L * DIM_GH * 2);
  _Float16* Wlv16  = (_Float16*)alloc((size_t)DIM_L * DIM_GH * 2);

  auto cvt = [&](const float* s, _Float16* d, int n) {
    f32_to_f16_kernel<<<(n + 255) / 256, 256, 0, stream>>>(s, d, n);
  };
  cvt(W_msg, Wmsg16, N_TYPES * DIMM * DIMD);
  cvt(W_ih,  Wih16,  384 * DIMM);
  cvt(W_hh,  Whh16,  384 * DIMD);
  cvt(W_n2g, Wn2g16, DIM_GH * DIMD);
  cvt(W_loc, Wloc16, DIM_L * DIM_GH);
  cvt(W_lv,  Wlv16,  DIM_L * DIM_GH);

  embed_kernel<<<(N_NODES * DIMD) / 256, 256, 0, stream>>>(node_types, emb, h, h16);

  for (int pass = 0; pass < N_PASSES; ++pass) {
    hipMemsetAsync(agg, 0, (size_t)N_NODES * DIMM * 4, stream);
    for (int t = 0; t < N_TYPES; ++t) {
      dim3 g(DIMM / 64, N_NODES / 64);
      gemm_wmma_kernel<<<g, 128, 0, stream>>>(
          h16, Wmsg16 + (size_t)t * DIMM * DIMD, b_msg + t * DIMM,
          msgs, DIMD, DIMM, 0, nullptr, nullptr);
      scatter_edges_kernel<<<(N_EDGES * 64) / 256, 256, 0, stream>>>(
          msgs, edge_src, edge_dst, edge_type, t, agg);
    }
    relu_to_f16_kernel<<<(N_NODES * DIMM) / 256, 256, 0, stream>>>(agg, m16, N_NODES * DIMM);
    {
      dim3 g(384 / 64, N_NODES / 64);
      gemm_wmma_kernel<<<g, 128, 0, stream>>>(m16, Wih16, b_ih, gi, DIMM, 384, 0, nullptr, nullptr);
      gemm_wmma_kernel<<<g, 128, 0, stream>>>(h16, Whh16, b_hh, gh, DIMD, 384, 0, nullptr, nullptr);
    }
    gru_kernel<<<(N_NODES * DIMD) / 256, 256, 0, stream>>>(gi, gh, h, h16);
  }

  // Readout
  gate_seg_kernel<<<(N_NODES + 255) / 256, 256, 0, stream>>>(h, W_gate, b_gate, ptr, gate, seg);
  hipMemsetAsync(hgraph, 0, (size_t)N_GRAPH * DIM_GH * 4, stream);
  {
    dim3 g(DIM_GH / 64, N_NODES / 64);
    gemm_wmma_kernel<<<g, 128, 0, stream>>>(h16, Wn2g16, b_n2g, hgraph, DIMD, DIM_GH, 1, seg, gate);
  }
  cvt(hgraph, hg16, N_GRAPH * DIM_GH);
  {
    dim3 g(DIM_L / 64, N_GRAPH / 64);
    gemm_wmma_kernel<<<g, 128, 0, stream>>>(hg16, Wloc16, b_loc, out, DIM_GH, DIM_L, 0, nullptr, nullptr);
    gemm_wmma_kernel<<<g, 128, 0, stream>>>(hg16, Wlv16, b_lv, out + (size_t)N_GRAPH * DIM_L,
                                            DIM_GH, DIM_L, 0, nullptr, nullptr);
  }
}
